// DLGPE_283467841759
// MI455X (gfx1250) — matrix operand order repaired and checked by reading, not compiled, not run on hardware
//
#include <hip/hip_runtime.h>
#include <hip/hip_bf16.h>
#include <stdint.h>

// ---------------------------------------------------------------------------
// MI455X (gfx1250): wave32, WMMA bf16 16x16x32 with fragment-packed LDS
// (2x ds_load_b128 per operand), TDM tensor_load_to_lds for attention k/v.
// ---------------------------------------------------------------------------

typedef __attribute__((ext_vector_type(16))) __bf16 v16bf;
typedef __attribute__((ext_vector_type(8)))  float  v8f;
typedef __attribute__((ext_vector_type(4)))  unsigned int u32x4;
typedef __attribute__((ext_vector_type(8)))  int i32x8;
typedef __attribute__((ext_vector_type(4)))  int i32x4;

#define DIMC   256
#define NPIX   16384       // 128*128
#define HWID   128
#define BATCH  8
#define WSZ    8
#define HEADS  4
#define SHIFT  3

#if defined(__AMDGCN__) && __has_builtin(__builtin_amdgcn_tensor_load_to_lds) && \
    __has_builtin(__builtin_amdgcn_s_wait_tensorcnt)
#define HAVE_TDM 1
#else
#define HAVE_TDM 0
#endif

static __device__ __forceinline__ __bf16 f2bf(float f) {
  uint32_t u = __builtin_bit_cast(uint32_t, f);
  u += 0x7FFFu + ((u >> 16) & 1u);          // round-to-nearest-even
  uint16_t h = (uint16_t)(u >> 16);
  return __builtin_bit_cast(__bf16, h);
}

static __device__ __forceinline__ float gelu_exact(float v) {
  return 0.5f * v * (1.0f + erff(v * 0.70710678118654752f));
}

// Packed-fragment addressing (this file's contiguous k order k = 16*(lane/16) + e, same for A and B):
//  A operand (16x32): element (m,k): lane = (k>>4)*16 + m, e = k&15
//  B operand (32x16): element (k,n): lane = (k>>4)*16 + n, e = k&15
// A packed subtile = 512 bf16 (lane-major, 16 elems/lane, 32B aligned).
static __device__ __forceinline__ int packA(int m, int k) {
  return ((k >> 4) * 16 + m) * 16 + (k & 15);
}
static __device__ __forceinline__ int packB(int k, int n) {
  return ((k >> 4) * 16 + n) * 16 + (k & 15);
}

static __device__ __forceinline__ v16bf frag(const __bf16* p) {
  return *(const v16bf*)(p + (threadIdx.x & 31) * 16);
}

static __device__ __forceinline__ v8f wmma_bf16(v16bf a, v16bf b, v8f c) {
  return __builtin_amdgcn_wmma_f32_16x16x32_bf16(false, a, false, b, (short)0, c,
                                                 false, false);
}

#if HAVE_TDM
// TDM: load one window tile (x=8, y=8 rows stride 128, z=256 planes stride
// 16384; f32 elements) from global into LDS [c][8][8] contiguous.
// amdgpu-toolchain (clang-23) 6-arg form: (g0, g1, g2, g3, g4, cpol).
static __device__ __forceinline__ void tdm_load_window(const float* gptr, void* lds) {
  uint64_t ga = (uint64_t)(uintptr_t)gptr;
  uint32_t la = (uint32_t)(uintptr_t)lds;
  u32x4 g0;
  g0[0] = 1u;                                   // count=1, user descriptor
  g0[1] = la;                                   // lds_addr
  g0[2] = (uint32_t)ga;                         // global_addr[31:0]
  g0[3] = (uint32_t)((ga >> 32) & 0x01FFFFFFu) | 0x80000000u;  // addr[56:32]|type=2
  i32x8 g1;
  g1[0] = 2 << 16;                              // data_size = 4B
  g1[1] = 128 << 16;                            // tensor_dim0 = 128
  g1[2] = 128 << 16;                            // tensor_dim1 = 128
  g1[3] = 8 << 16;                              // tile_dim0 = 8
  g1[4] = 8 | (256 << 16);                      // tile_dim1 = 8, tile_dim2 = 256
  g1[5] = 128;                                  // tensor_dim0_stride = 128
  g1[6] = 16384 << 16;                          // tensor_dim1_stride = 16384
  g1[7] = 0;
  i32x4 g2;
  g2[0] = 256;                                  // tensor_dim2 = 256
  g2[1] = 0;
  g2[2] = 16384;                                // tensor_dim2_stride
  g2[3] = 0;
  i32x4 g3;
  g3[0] = 0; g3[1] = 0; g3[2] = 0; g3[3] = 0;
  i32x8 g4;
  g4[0] = 0; g4[1] = 0; g4[2] = 0; g4[3] = 0;
  g4[4] = 0; g4[5] = 0; g4[6] = 0; g4[7] = 0;
  __builtin_amdgcn_tensor_load_to_lds(g0, g1, g2, g3, g4, 0);
}
#endif

// ---------------------------------------------------------------------------
// 1) per-(b,c) spatial mean  -> means[b*256+c]
// ---------------------------------------------------------------------------
__global__ __launch_bounds__(256) void k_plane_mean(const float* __restrict__ X,
                                                    float* __restrict__ means) {
  __shared__ float red[256];
  int bc = blockIdx.x;
  const float* p = X + (size_t)bc * NPIX;
  float s = 0.f;
#pragma unroll 4
  for (int i = 0; i < 64; ++i) s += p[threadIdx.x + i * 256];
  red[threadIdx.x] = s;
  __syncthreads();
  for (int off = 128; off > 0; off >>= 1) {
    if (threadIdx.x < off) red[threadIdx.x] += red[threadIdx.x + off];
    __syncthreads();
  }
  if (threadIdx.x == 0) means[bc] = red[0] * (1.0f / NPIX);
}

// ---------------------------------------------------------------------------
// 2) gating MLP: pooled -> 1x1(256->64)+LN+gelu -> 1x1(64->3)+LN+softmax
// ---------------------------------------------------------------------------
__global__ __launch_bounds__(64) void k_gate(const float* __restrict__ means,
                                             const float* __restrict__ w0, const float* __restrict__ b0,
                                             const float* __restrict__ g0, const float* __restrict__ be0,
                                             const float* __restrict__ w1, const float* __restrict__ b1,
                                             const float* __restrict__ g1, const float* __restrict__ be1,
                                             float* __restrict__ cw) {
  __shared__ float pooled[DIMC];
  __shared__ float h1[64];
  __shared__ float h2[3];
  int b = blockIdx.x, t = threadIdx.x;
  for (int i = 0; i < 4; ++i) pooled[t + 64 * i] = means[b * DIMC + t + 64 * i];
  __syncthreads();
  float acc = b0[t];
  for (int ci = 0; ci < DIMC; ++ci) acc += w0[t * DIMC + ci] * pooled[ci];
  h1[t] = acc;
  __syncthreads();
  float mn = 0.f, vr = 0.f;
  for (int k = 0; k < 64; ++k) mn += h1[k];
  mn *= (1.f / 64.f);
  for (int k = 0; k < 64; ++k) { float d = h1[k] - mn; vr += d * d; }
  vr *= (1.f / 64.f);
  float gl = gelu_exact((h1[t] - mn) * rsqrtf(vr + 1e-6f) * g0[t] + be0[t]);
  __syncthreads();
  h1[t] = gl;
  __syncthreads();
  if (t < 3) {
    float a2 = b1[t];
    for (int j = 0; j < 64; ++j) a2 += w1[t * 64 + j] * h1[j];
    h2[t] = a2;
  }
  __syncthreads();
  if (t == 0) {
    float m2 = (h2[0] + h2[1] + h2[2]) * (1.f / 3.f);
    float v2 = 0.f;
    for (int k = 0; k < 3; ++k) { float d = h2[k] - m2; v2 += d * d; }
    v2 *= (1.f / 3.f);
    float nv[3], mx = -1e30f, sm = 0.f;
    for (int k = 0; k < 3; ++k) {
      nv[k] = (h2[k] - m2) * rsqrtf(v2 + 1e-6f) * g1[k] + be1[k];
      mx = fmaxf(mx, nv[k]);
    }
    for (int k = 0; k < 3; ++k) { nv[k] = __expf(nv[k] - mx); sm += nv[k]; }
    for (int k = 0; k < 3; ++k) cw[b * 3 + k] = 0.25f + 0.25f * nv[k] / sm;
  }
}

// ---------------------------------------------------------------------------
// 3) 1x1 conv as GEMM (M=768,K=256,N=B*16384), packed-fragment LDS staging
// ---------------------------------------------------------------------------
__global__ __launch_bounds__(256) void k_gemm_qkv(const float* __restrict__ X,
                                                  const float* __restrict__ W,
                                                  const float* __restrict__ bias,
                                                  float* __restrict__ Y,
                                                  const float* __restrict__ scale,
                                                  int scaleIdx) {
  __shared__ __bf16 Ap[4 * 512];   // 4 m-subtiles, packed
  __shared__ __bf16 Bp[4 * 512];   // 4 n-subtiles, packed
  int p0 = blockIdx.x * 64;
  int m0 = blockIdx.y * 64;
  int b  = p0 >> 14;
  int pix0 = p0 & (NPIX - 1);
  float s = scale ? scale[b * 3 + scaleIdx] : 1.0f;

  int wave = threadIdx.x >> 5;
  int t0 = wave * 2;
  v8f acc[2];
#pragma unroll
  for (int j = 0; j < 2; ++j)
#pragma unroll
    for (int r = 0; r < 8; ++r) acc[j][r] = 0.f;

  const size_t xb = (size_t)b * DIMC * NPIX;
  for (int kc = 0; kc < DIMC; kc += 32) {
    for (int i = threadIdx.x; i < 64 * 32; i += 256) {
      int mi = i >> 5, ki = i & 31;
      Ap[(mi >> 4) * 512 + packA(mi & 15, ki)] =
          f2bf(W[(size_t)(m0 + mi) * DIMC + kc + ki]);
    }
    for (int i = threadIdx.x; i < 32 * 64; i += 256) {
      int ki = i >> 6, ni = i & 63;
      Bp[(ni >> 4) * 512 + packB(ki, ni & 15)] =
          f2bf(X[xb + (size_t)(kc + ki) * NPIX + pix0 + ni] * s);
    }
    __syncthreads();
#pragma unroll
    for (int j = 0; j < 2; ++j) {
      int tid = t0 + j;
      int mt = tid >> 2, nt = tid & 3;
      acc[j] = wmma_bf16(frag(Ap + mt * 512), frag(Bp + nt * 512), acc[j]);
    }
    __syncthreads();
  }
  int lane = threadIdx.x & 31;
  int half = (lane >= 16) ? 8 : 0;
  int nloc = lane & 15;
#pragma unroll
  for (int j = 0; j < 2; ++j) {
    int tid = t0 + j;
    int mt = tid >> 2, nt = tid & 3;
#pragma unroll
    for (int r = 0; r < 8; ++r) {
      int row = m0 + mt * 16 + r + half;
      int pix = pix0 + nt * 16 + nloc;
      Y[(size_t)b * 768 * NPIX + (size_t)row * NPIX + pix] = acc[j][r] + bias[row];
    }
  }
}

// ---------------------------------------------------------------------------
// 4) fused window attention. TDM loads k/v window tiles (in-bounds, no roll),
//    q loaded manually (may wrap under shift). All operands repacked to
//    fragment-packed LDS; softmax P written directly in packed B layout.
// ---------------------------------------------------------------------------
__global__ __launch_bounds__(256) void k_window_attn(const float* __restrict__ qkv,
                                                     const float* __restrict__ lng,
                                                     const float* __restrict__ lnb,
                                                     const float* __restrict__ rpb,
                                                     float* __restrict__ out,
                                                     int shift, int accumulate) {
  extern __shared__ char smem[];
  float*  qf   = (float*)smem;                    //  64 KB : q window f32
  float*  R1   = (float*)(smem + 65536);          //  64 KB : TDM raw k/v f32
  __bf16* qs_p = (__bf16*)(smem + 131072);        //  32 KB packed B operands
  __bf16* ks_p = (__bf16*)(smem + 163840);        //  32 KB packed A operands
  __bf16* vs_p = (__bf16*)(smem + 196608);        //  32 KB packed A operands
  __bf16* ps_p = (__bf16*)(smem + 229376);        //  32 KB packed B operands
  float*  mu   = (float*)(smem + 262144);
  float*  rs   = (float*)(smem + 262400);

  int bi = blockIdx.x;
  int b   = bi >> 8;
  int wy0 = ((bi >> 4) & 15) * WSZ;
  int wx0 = (bi & 15) * WSZ;
  const size_t base = (size_t)b * 768 * NPIX;
  const int wave = threadIdx.x >> 5;
  const int lane = threadIdx.x & 31;

  // --- phase 0: DMA k tile; overlap with q global loads (rolled by -shift)
#if HAVE_TDM
  if (wave == 0)
    tdm_load_window(qkv + base + (size_t)256 * NPIX + wy0 * HWID + wx0, R1);
#endif
  for (int i = 0; i < 64; ++i) {
    int idx = threadIdx.x + i * 256;
    int c = idx >> 6, n = idx & 63;
    int gy = wy0 + (n >> 3), gx = wx0 + (n & 7);
    int qy = (gy + shift) & (HWID - 1), qx = (gx + shift) & (HWID - 1);
    qf[idx] = qkv[base + (size_t)c * NPIX + qy * HWID + qx];
  }
#if HAVE_TDM
  if (wave == 0) __builtin_amdgcn_s_wait_tensorcnt(0);
#else
  for (int i = 0; i < 64; ++i) {
    int idx = threadIdx.x + i * 256;
    int c = idx >> 6, n = idx & 63;
    int gy = wy0 + (n >> 3), gx = wx0 + (n & 7);
    R1[idx] = qkv[base + (size_t)(256 + c) * NPIX + gy * HWID + gx];
  }
#endif
  __syncthreads();

  // --- phase 1: repack k -> ks_p (A operand of S = K^T x Qn)
  for (int i = 0; i < 64; ++i) {
    int idx = threadIdx.x + i * 256;
    int c = idx >> 6, n = idx & 63;
    int h = c >> 6, kkg = c & 63;
    int st = ((h * 4 + (n >> 4)) * 2 + (kkg >> 5));
    ks_p[st * 512 + packA(n & 15, kkg & 31)] = f2bf(R1[idx]);
  }
  __syncthreads();   // R1 free

  // --- phase 2: DMA v tile; overlap with LayerNorm(q) + packing q
#if HAVE_TDM
  if (wave == 0)
    tdm_load_window(qkv + base + (size_t)512 * NPIX + wy0 * HWID + wx0, R1);
#else
  for (int i = 0; i < 64; ++i) {
    int idx = threadIdx.x + i * 256;
    int c = idx >> 6, n = idx & 63;
    int gy = wy0 + (n >> 3), gx = wx0 + (n & 7);
    R1[idx] = qkv[base + (size_t)(512 + c) * NPIX + gy * HWID + gx];
  }
#endif
  if (threadIdx.x < 64) {
    int n = threadIdx.x;
    float m = 0.f;
    for (int c = 0; c < DIMC; ++c) m += qf[c * 64 + n];
    m *= (1.f / DIMC);
    float v = 0.f;
    for (int c = 0; c < DIMC; ++c) { float d = qf[c * 64 + n] - m; v += d * d; }
    mu[n] = m;
    rs[n] = rsqrtf(v * (1.f / DIMC) + 1e-6f);
  }
  __syncthreads();
  for (int i = 0; i < 64; ++i) {
    int idx = threadIdx.x + i * 256;
    int c = idx >> 6, n = idx & 63;
    float v = (qf[idx] - mu[n]) * rs[n] * lng[c] + lnb[c];
    int h = c >> 6, kkg = c & 63;
    int st = ((h * 4 + (n >> 4)) * 2 + (kkg >> 5));
    qs_p[st * 512 + packB(kkg & 31, n & 15)] = f2bf(v);
  }
#if HAVE_TDM
  if (wave == 0) __builtin_amdgcn_s_wait_tensorcnt(0);
#endif
  __syncthreads();

  // --- phase 3: repack v -> vs_p (A operand of O = V x P; K-dim = pixel m)
  for (int i = 0; i < 64; ++i) {
    int idx = threadIdx.x + i * 256;
    int c = idx >> 6, n = idx & 63;
    int h = c >> 6, ddg = c & 63;
    int st = ((h * 4 + (ddg >> 4)) * 2 + (n >> 5));
    vs_p[st * 512 + packA(ddg & 15, n & 31)] = f2bf(R1[idx]);
  }
  __syncthreads();

  const int head = wave >> 1;
  const int np   = wave & 1;           // owns n-tiles {2np, 2np+1}
  const int half = (lane >= 16) ? 8 : 0;
  const int nloc = lane & 15;

  // --- S'[m][n] (softmax dim = WMMA M)
  v8f acc[2][4];
#pragma unroll
  for (int j = 0; j < 2; ++j)
#pragma unroll
    for (int mt = 0; mt < 4; ++mt)
#pragma unroll
      for (int r = 0; r < 8; ++r) acc[j][mt][r] = 0.f;

#pragma unroll
  for (int kcI = 0; kcI < 2; ++kcI) {
    v16bf bfr[2];
#pragma unroll
    for (int j = 0; j < 2; ++j)
      bfr[j] = frag(qs_p + ((head * 4 + 2 * np + j) * 2 + kcI) * 512);
#pragma unroll
    for (int mt = 0; mt < 4; ++mt) {
      v16bf a = frag(ks_p + ((head * 4 + mt) * 2 + kcI) * 512);
#pragma unroll
      for (int j = 0; j < 2; ++j) acc[j][mt] = wmma_bf16(a, bfr[j], acc[j][mt]);
    }
  }

  // --- bias + softmax over m, write P directly packed (B operand layout)
#pragma unroll
  for (int j = 0; j < 2; ++j) {
    int n = (2 * np + j) * 16 + nloc;
    int i1 = n >> 3, j1 = n & 7;
    float mx = -1e30f;
#pragma unroll
    for (int mt = 0; mt < 4; ++mt)
#pragma unroll
      for (int r = 0; r < 8; ++r) {
        int m = mt * 16 + r + half;
        int i2 = m >> 3, j2 = m & 7;
        int ridx = (i1 - i2 + 7) * 15 + (j1 - j2 + 7);
        float v = acc[j][mt][r] * 0.125f + rpb[ridx * HEADS + head];
        acc[j][mt][r] = v;
        mx = fmaxf(mx, v);
      }
    mx = fmaxf(mx, __shfl_xor(mx, 16, 32));
    float sum = 0.f;
#pragma unroll
    for (int mt = 0; mt < 4; ++mt)
#pragma unroll
      for (int r = 0; r < 8; ++r) {
        float e = __expf(acc[j][mt][r] - mx);
        acc[j][mt][r] = e;
        sum += e;
      }
    sum += __shfl_xor(sum, 16, 32);
    float inv = 1.f / sum;
#pragma unroll
    for (int mt = 0; mt < 4; ++mt) {
      // element (m = mt*16+r+half, n): st = (head*4+nt)*2 + (m>>5),
      // lane' = ((m>>4)&1)*16 + nloc, e = r+half  -> 8 contiguous bf16
      __bf16* dst = ps_p + ((head * 4 + 2 * np + j) * 2 + (mt >> 1)) * 512 +
                    ((mt & 1) * 16 + nloc) * 16 + half;
#pragma unroll
      for (int r = 0; r < 8; ++r) dst[r] = f2bf(acc[j][mt][r] * inv);
    }
  }
  __syncthreads();

  // --- O'[dd][n] = V x P
  v8f oac[2][4];
#pragma unroll
  for (int j = 0; j < 2; ++j)
#pragma unroll
    for (int dt = 0; dt < 4; ++dt)
#pragma unroll
      for (int r = 0; r < 8; ++r) oac[j][dt][r] = 0.f;

#pragma unroll
  for (int kcI = 0; kcI < 2; ++kcI) {
    v16bf bfr[2];
#pragma unroll
    for (int j = 0; j < 2; ++j)
      bfr[j] = frag(ps_p + ((head * 4 + 2 * np + j) * 2 + kcI) * 512);
#pragma unroll
    for (int dt = 0; dt < 4; ++dt) {
      v16bf a = frag(vs_p + ((head * 4 + dt) * 2 + kcI) * 512);
#pragma unroll
      for (int j = 0; j < 2; ++j) oac[j][dt] = wmma_bf16(a, bfr[j], oac[j][dt]);
    }
  }

  // --- window_reverse + roll-back(+shift) store
#pragma unroll
  for (int j = 0; j < 2; ++j) {
    int n = (2 * np + j) * 16 + nloc;
    int gy = wy0 + (n >> 3), gx = wx0 + (n & 7);
    int oy = (gy + shift) & (HWID - 1), ox = (gx + shift) & (HWID - 1);
#pragma unroll
    for (int dt = 0; dt < 4; ++dt)
#pragma unroll
      for (int r = 0; r < 8; ++r) {
        int c = head * 64 + dt * 16 + r + half;
        size_t addr = (size_t)b * DIMC * NPIX + (size_t)c * NPIX + oy * HWID + ox;
        float v = oac[j][dt][r];
        if (accumulate) out[addr] += v; else out[addr] = v;
      }
  }
}

// ---------------------------------------------------------------------------
// 5) depthwise 3x3 (pad==dil), optional per-batch scale on input
// ---------------------------------------------------------------------------
__global__ __launch_bounds__(256) void k_dwconv(const float* __restrict__ X,
                                                const float* __restrict__ W,
                                                const float* __restrict__ bias,
                                                float* __restrict__ Y,
                                                int dil,
                                                const float* __restrict__ scale,
                                                int scaleIdx) {
  size_t idx = (size_t)blockIdx.x * 256 + threadIdx.x;
  int x_ = idx & 127;
  int y_ = (idx >> 7) & 127;
  int c  = (idx >> 14) & 255;
  int b  = (int)(idx >> 22);
  float s = scale ? scale[b * 3 + scaleIdx] : 1.0f;
  const float* xp = X + ((size_t)b * DIMC + c) * NPIX;
  __builtin_prefetch(xp + (y_ + dil < HWID ? (y_ + dil) * HWID + x_ : 0), 0, 1);
  float acc = bias[c];
#pragma unroll
  for (int ky = 0; ky < 3; ++ky) {
    int yy = y_ + (ky - 1) * dil;
    if (yy < 0 || yy >= HWID) continue;
#pragma unroll
    for (int kx = 0; kx < 3; ++kx) {
      int xx = x_ + (kx - 1) * dil;
      if (xx < 0 || xx >= HWID) continue;
      acc += xp[yy * HWID + xx] * s * W[c * 9 + ky * 3 + kx];
    }
  }
  Y[idx] = acc;
}

// ---------------------------------------------------------------------------
// 6) mkc: KSxKS conv (64ch) as implicit GEMM + LN + GELU. Weights and the
//    im2col matrix are both materialized in packed-fragment LDS once.
// ---------------------------------------------------------------------------
template <int KS>
__global__ __launch_bounds__(256) void k_conv_ln_gelu(const float* __restrict__ X,
                                                      int xBatchStride,
                                                      const float* __restrict__ W,
                                                      const float* __restrict__ bias,
                                                      const float* __restrict__ g,
                                                      const float* __restrict__ be,
                                                      float* __restrict__ Y,
                                                      int yBatchStride, int yChanOff,
                                                      int accumulate,
                                                      const float* __restrict__ scale) {
  constexpr int K   = 64 * KS * KS;
  constexpr int NK  = K / 32;
  constexpr int PAD = KS / 2;
  constexpr int PW  = 8 + 2 * PAD;
  extern __shared__ char smem[];
  __bf16* Wp    = (__bf16*)smem;                               // 4*NK*512
  __bf16* patch = (__bf16*)(smem + (size_t)64 * K * 2);        // 64*PW*PW
  __bf16* Bim   = (__bf16*)(smem + (size_t)64 * K * 2 + 64 * PW * PW * 2);
  float*  outf  = (float*)(smem + (size_t)128 * K * 2 + 64 * PW * PW * 2);
  float*  mu    = outf + 4096;
  float*  rs    = mu + 64;

  int bi = blockIdx.x;
  int b  = bi >> 8;
  int ty = ((bi >> 4) & 15) * 8;
  int tx = (bi & 15) * 8;
  float s = scale ? scale[b * 3 + 0] : 1.0f;

  for (int i = threadIdx.x; i < 64 * K; i += 256) {
    int co = i / K, k = i % K;
    Wp[((co >> 4) * NK + (k >> 5)) * 512 + packA(co & 15, k & 31)] = f2bf(W[i]);
  }
  for (int i = threadIdx.x; i < 64 * PW * PW; i += 256) {
    int ci = i / (PW * PW);
    int rr = (i / PW) % PW;
    int cc = i % PW;
    int gy = ty + rr - PAD, gx = tx + cc - PAD;
    float v = 0.f;
    if (gy >= 0 && gy < HWID && gx >= 0 && gx < HWID)
      v = X[(size_t)b * xBatchStride + (size_t)ci * NPIX + gy * HWID + gx] * s;
    patch[i] = f2bf(v);
  }
  __syncthreads();
  // packed im2col B (K x 64 pixels)
  for (int i = threadIdx.x; i < K * 64; i += 256) {
    int k = i >> 6, npix = i & 63;
    int ci  = k / (KS * KS);
    int rem = k % (KS * KS);
    int ky = rem / KS, kx = rem % KS;
    int py = npix >> 3, px = npix & 7;
    Bim[((npix >> 4) * NK + (k >> 5)) * 512 + packB(k & 31, npix & 15)] =
        patch[ci * PW * PW + (py + ky) * PW + (px + kx)];
  }
  __syncthreads();

  int wave = threadIdx.x >> 5;
  int mt  = wave & 3;
  int ntp = wave >> 2;
  int lane = threadIdx.x & 31;
  int half = (lane >= 16) ? 8 : 0;
  int nloc = lane & 15;

  v8f acc[2];
#pragma unroll
  for (int j = 0; j < 2; ++j)
#pragma unroll
    for (int r = 0; r < 8; ++r) acc[j][r] = 0.f;

  for (int kc = 0; kc < NK; ++kc) {
    v16bf a = frag(Wp + (mt * NK + kc) * 512);
#pragma unroll
    for (int j = 0; j < 2; ++j)
      acc[j] = wmma_bf16(a, frag(Bim + ((2 * ntp + j) * NK + kc) * 512), acc[j]);
  }

#pragma unroll
  for (int j = 0; j < 2; ++j) {
    int npix = (2 * ntp + j) * 16 + nloc;
#pragma unroll
    for (int r = 0; r < 8; ++r) {
      int c = mt * 16 + r + half;
      outf[c * 64 + npix] = acc[j][r] + bias[c];
    }
  }
  __syncthreads();
  if (threadIdx.x < 64) {
    int npix = threadIdx.x;
    float m = 0.f;
    for (int c = 0; c < 64; ++c) m += outf[c * 64 + npix];
    m *= (1.f / 64.f);
    float v = 0.f;
    for (int c = 0; c < 64; ++c) { float d = outf[c * 64 + npix] - m; v += d * d; }
    mu[npix] = m;
    rs[npix] = rsqrtf(v * (1.f / 64.f) + 1e-6f);
  }
  __syncthreads();
  for (int i = threadIdx.x; i < 4096; i += 256) {
    int c = i >> 6, npix = i & 63;
    float v = (outf[i] - mu[npix]) * rs[npix] * g[c] + be[c];
    v = gelu_exact(v);
    size_t addr = (size_t)b * yBatchStride + (size_t)(yChanOff + c) * NPIX +
                  (size_t)(ty + (npix >> 3)) * HWID + tx + (npix & 7);
    if (accumulate) Y[addr] += v; else Y[addr] = v;
  }
}

// ---------------------------------------------------------------------------
// host-side orchestration
// ---------------------------------------------------------------------------
extern "C" void kernel_launch(void* const* d_in, const int* in_sizes, int n_in,
                              void* d_out, int out_size, void* d_ws, size_t ws_size,
                              hipStream_t stream) {
  const float* x     = (const float*)d_in[0];
  const float* ca0w  = (const float*)d_in[1];
  const float* ca0b  = (const float*)d_in[2];
  const float* ca0g  = (const float*)d_in[3];
  const float* ca0be = (const float*)d_in[4];
  const float* ca1w  = (const float*)d_in[5];
  const float* ca1b  = (const float*)d_in[6];
  const float* ca1g  = (const float*)d_in[7];
  const float* ca1be = (const float*)d_in[8];
  const float* wqkvW = (const float*)d_in[9];
  const float* wqkvB = (const float*)d_in[10];
  const float* sw0w  = (const float*)d_in[11];
  const float* sw0b  = (const float*)d_in[12];
  const float* sw1w  = (const float*)d_in[13];
  const float* sw1b  = (const float*)d_in[14];
  const float* sw2w  = (const float*)d_in[15];
  const float* sw2b  = (const float*)d_in[16];
  const float* lng   = (const float*)d_in[17];
  const float* lnbe  = (const float*)d_in[18];
  const float* rpb   = (const float*)d_in[19];
  const int mkcBase[7] = {20, 24, 28, 32, 36, 40, 44};

  float* out = (float*)d_out;
  char* ws = (char*)d_ws;
  float* means = (float*)(ws);
  float* cw    = (float*)(ws + 8192);
  float* qkv   = (float*)(ws + 16384);
  size_t offA  = 16384 + (size_t)BATCH * 768 * NPIX * 4;
  float* bufA  = (float*)(ws + offA);
  float* bufB  = (float*)(ws + offA + (size_t)BATCH * DIMC * NPIX * 4);

  const size_t smemAttn = 262656;
  const size_t smemC3 = (size_t)128 * 576 * 2 + 64 * 100 * 2 + 4096 * 4 + 512;
  const size_t smemC1 = (size_t)128 * 64 * 2 + 64 * 64 * 2 + 4096 * 4 + 512;

  k_plane_mean<<<BATCH * DIMC, 256, 0, stream>>>(x, means);
  k_gate<<<BATCH, 64, 0, stream>>>(means, ca0w, ca0b, ca0g, ca0be,
                                   ca1w, ca1b, ca1g, ca1be, cw);

  // plain window branch (scale cw[:,1] fused into GEMM)
  k_gemm_qkv<<<dim3(2048, 12), 256, 0, stream>>>(x, wqkvW, wqkvB, qkv, cw, 1);
  k_window_attn<<<2048, 256, smemAttn, stream>>>(qkv, lng, lnbe, rpb, out, 0, 0);

  // shifted window branch (scale cw[:,2] on first depthwise)
  k_dwconv<<<(BATCH * DIMC * NPIX) / 256, 256, 0, stream>>>(x, sw0w, sw0b, bufA, 1, cw, 2);
  k_dwconv<<<(BATCH * DIMC * NPIX) / 256, 256, 0, stream>>>(bufA, sw1w, sw1b, bufB, 2, nullptr, 0);
  k_gemm_qkv<<<dim3(2048, 12), 256, 0, stream>>>(bufB, sw2w, sw2b, qkv, nullptr, 0);
  k_window_attn<<<2048, 256, smemAttn, stream>>>(qkv, lng, lnbe, rpb, out, SHIFT, 1);

  // multi-kernel conv branch (scale cw[:,0] on each split's first conv)
  auto L = [&](int li, int idx) { return (const float*)d_in[mkcBase[li] + idx]; };
  const int xBS = DIMC * NPIX, iBS = 64 * NPIX;
  k_conv_ln_gelu<3><<<2048, 256, smemC3, stream>>>(x + 0 * 64 * NPIX, xBS,
      L(0,0), L(0,1), L(0,2), L(0,3), bufA, iBS, 0, 0, cw);
  k_conv_ln_gelu<3><<<2048, 256, smemC3, stream>>>(bufA, iBS,
      L(1,0), L(1,1), L(1,2), L(1,3), bufB, iBS, 0, 0, nullptr);
  k_conv_ln_gelu<3><<<2048, 256, smemC3, stream>>>(bufB, iBS,
      L(2,0), L(2,1), L(2,2), L(2,3), out, xBS, 0, 1, nullptr);
  k_conv_ln_gelu<3><<<2048, 256, smemC3, stream>>>(x + 1 * 64 * NPIX, xBS,
      L(3,0), L(3,1), L(3,2), L(3,3), bufA, iBS, 0, 0, cw);
  k_conv_ln_gelu<3><<<2048, 256, smemC3, stream>>>(bufA, iBS,
      L(4,0), L(4,1), L(4,2), L(4,3), out, xBS, 64, 1, nullptr);
  k_conv_ln_gelu<3><<<2048, 256, smemC3, stream>>>(x + 2 * 64 * NPIX, xBS,
      L(5,0), L(5,1), L(5,2), L(5,3), out, xBS, 128, 1, cw);
  k_conv_ln_gelu<1><<<2048, 256, smemC1, stream>>>(x + 3 * 64 * NPIX, xBS,
      L(6,0), L(6,1), L(6,2), L(6,3), out, xBS, 192, 1, cw);

  (void)in_sizes; (void)n_in; (void)out_size; (void)ws_size;
}
